// Linear_int8_T_21663815041268
// MI455X (gfx1250) — compile-verified, matrix operand order checked by reading, not run on hardware
//
#include <hip/hip_runtime.h>
#include <stdint.h>

typedef __attribute__((ext_vector_type(8))) int v8i;

#define N_DIM 128
#define K_DIM 1024
#define M_DIM 1024

// ---------------------------------------------------------------------------
// Pass 1: quantize f32 -> signed int8 (packed 4-per-dword), q = clip(rint(v/s))
// ---------------------------------------------------------------------------
__global__ void quantize_pack_kernel(const float* __restrict__ src,
                                     int* __restrict__ dst,
                                     int nWords, float invScale) {
  int i = blockIdx.x * blockDim.x + threadIdx.x;
  if (i >= nWords) return;
  const float4 v = reinterpret_cast<const float4*>(src)[i];
  int q0 = (int)rintf(fminf(fmaxf(v.x * invScale, -128.0f), 127.0f));
  int q1 = (int)rintf(fminf(fmaxf(v.y * invScale, -128.0f), 127.0f));
  int q2 = (int)rintf(fminf(fmaxf(v.z * invScale, -128.0f), 127.0f));
  int q3 = (int)rintf(fminf(fmaxf(v.w * invScale, -128.0f), 127.0f));
  dst[i] = (q0 & 255) | ((q1 & 255) << 8) | ((q2 & 255) << 16) | (q3 << 24);
}

// ---------------------------------------------------------------------------
// Pass 2: int8 GEMM via V_WMMA_I32_16X16X64_IU8 (signed x signed).
// Block = 128 threads = 4 wave32s. Each wave: 16(N) x 64(M) output strip.
// A (x tile, 16x64 i8) staged into LDS via GLOBAL_LOAD_ASYNC_TO_LDS_B64
// (ASYNCcnt path) and shared by all 4 waves.
// ---------------------------------------------------------------------------
__global__ __launch_bounds__(128) void wmma_i8_gemm_kernel(
    const uint8_t* __restrict__ xq,   // N x K  int8 (signed bit pattern)
    const uint8_t* __restrict__ wq,   // M x K  int8
    const float* __restrict__ bias,   // M
    float* __restrict__ out,          // N x M
    float scale) {
  __shared__ alignas(16) uint8_t sA[16 * 64];

  const int lane = threadIdx.x & 31;
  const int wave = threadIdx.x >> 5;
  const int row  = lane & 15;   // matrix row owned by this lane (A/B frag)
  const int half = lane >> 4;   // K-chunk selector per ISA 8-bit layout

  const int n0 = blockIdx.y * 16;                 // output row block
  const int m0 = blockIdx.x * 256 + wave * 64;    // output col strip (per wave)

  v8i acc[4];
  #pragma unroll
  for (int t = 0; t < 4; ++t) acc[t] = v8i{};

  // cooperative LDS staging coordinates (128 threads x 8B = 1KB tile)
  const int srow = threadIdx.x >> 3;        // 0..15
  const int scol = (threadIdx.x & 7) * 8;   // 0,8,...,56

  // LDS byte offset of this thread's staging slot: flat shared address
  // truncated to 32 bits (LDS aperture mapping, ISA 10.2).
  const uint32_t lds_dst = (uint32_t)(uintptr_t)&sA[srow * 64 + scol];

  for (int k = 0; k < K_DIM; k += 64) {
    // async stage of the 16x64 int8 x-tile straight into LDS (no VGPR trip)
    {
      const uint64_t gsrc =
          (uint64_t)(uintptr_t)&xq[(size_t)(n0 + srow) * K_DIM + k + scol];
      asm volatile("global_load_async_to_lds_b64 %0, %1, off"
                   :: "v"(lds_dst), "v"(gsrc) : "memory");
    }
    asm volatile("s_wait_asynccnt 0x0" ::: "memory");
    __syncthreads();

    // A fragment: lane L holds row (L&15); VGPR v holds K =
    //   (v>>1)*16 + (v&1)*4 + (L>>4)*8  .. +3   (8-bit 16x64 ISA layout)
    v8i a;
    {
      const uint8_t* ap = &sA[row * 64 + half * 8];
      uint64_t d0 = *reinterpret_cast<const uint64_t*>(ap + 0);
      uint64_t d1 = *reinterpret_cast<const uint64_t*>(ap + 16);
      uint64_t d2 = *reinterpret_cast<const uint64_t*>(ap + 32);
      uint64_t d3 = *reinterpret_cast<const uint64_t*>(ap + 48);
      a[0] = (int)(uint32_t)d0; a[1] = (int)(uint32_t)(d0 >> 32);
      a[2] = (int)(uint32_t)d1; a[3] = (int)(uint32_t)(d1 >> 32);
      a[4] = (int)(uint32_t)d2; a[5] = (int)(uint32_t)(d2 >> 32);
      a[6] = (int)(uint32_t)d3; a[7] = (int)(uint32_t)(d3 >> 32);
    }

    // 4 B fragments (64x16 = transpose of 16 weight rows x 64 K, same
    // per-lane layout as A applied to weight rows), loaded from global.
    #pragma unroll
    for (int t = 0; t < 4; ++t) {
      const uint8_t* bp =
          &wq[(size_t)(m0 + t * 16 + row) * K_DIM + k + half * 8];
      uint64_t d0 = *reinterpret_cast<const uint64_t*>(bp + 0);
      uint64_t d1 = *reinterpret_cast<const uint64_t*>(bp + 16);
      uint64_t d2 = *reinterpret_cast<const uint64_t*>(bp + 32);
      uint64_t d3 = *reinterpret_cast<const uint64_t*>(bp + 48);
      v8i b;
      b[0] = (int)(uint32_t)d0; b[1] = (int)(uint32_t)(d0 >> 32);
      b[2] = (int)(uint32_t)d1; b[3] = (int)(uint32_t)(d1 >> 32);
      b[4] = (int)(uint32_t)d2; b[5] = (int)(uint32_t)(d2 >> 32);
      b[6] = (int)(uint32_t)d3; b[7] = (int)(uint32_t)(d3 >> 32);

      // (sgn_a, A, sgn_b, B, C, reuse_a, reuse_b) -> v_wmma_i32_16x16x64_iu8
      acc[t] = __builtin_amdgcn_wmma_i32_16x16x64_iu8(
          /*sgn_a=*/true, a, /*sgn_b=*/true, b, acc[t],
          /*reuse_a=*/false, /*reuse_b=*/false);
    }
    __syncthreads();
  }

  // Epilogue: 16x16 i32 C/D layout — lane = column (m), VGPR g = row (n),
  // upper lane half owns rows 8..15.
  const int col = lane & 15;
  #pragma unroll
  for (int t = 0; t < 4; ++t) {
    const int m = m0 + t * 16 + col;
    const float bb = bias[m];
    #pragma unroll
    for (int g = 0; g < 8; ++g) {
      const int n = n0 + half * 8 + g;
      out[(size_t)n * M_DIM + m] = (float)acc[t][g] * scale + bb;
    }
  }
}

// ---------------------------------------------------------------------------
extern "C" void kernel_launch(void* const* d_in, const int* in_sizes, int n_in,
                              void* d_out, int out_size, void* d_ws, size_t ws_size,
                              hipStream_t stream) {
  const float* x    = (const float*)d_in[0];   // 128 x 1024
  const float* w    = (const float*)d_in[1];   // 1024 x 1024
  const float* bias = (const float*)d_in[2];   // 1024
  // d_in[3] (lut) intentionally unused: lut[i][j] == (i-128)*(j-128) exactly,
  // so the gather+sum is a signed-int8 GEMM computed exactly by IU8 WMMA.
  float* out = (float*)d_out;

  uint8_t* xq = (uint8_t*)d_ws;                      // 128*1024  int8
  uint8_t* wq = xq + (size_t)N_DIM * K_DIM;          // 1024*1024 int8

  const int xWords = N_DIM * K_DIM / 4;
  quantize_pack_kernel<<<(xWords + 255) / 256, 256, 0, stream>>>(
      x, (int*)xq, xWords, 127.0f / 3.0f);

  const int wWords = M_DIM * K_DIM / 4;
  quantize_pack_kernel<<<(wWords + 255) / 256, 256, 0, stream>>>(
      w, (int*)wq, wWords, 127.0f / 0.5f);

  const float scale = (3.0f / 127.0f) * (0.5f / 127.0f);
  dim3 grid(M_DIM / 256, N_DIM / 16);   // 4 x 8 blocks, 4 waves each
  wmma_i8_gemm_kernel<<<grid, 128, 0, stream>>>(xq, wq, bias, out, scale);
}
